// PLVConnectivity_80719615361130
// MI455X (gfx1250) — matrix operand order repaired and checked by reading, not compiled, not run on hardware
//
#include <hip/hip_runtime.h>
#include <cstdint>
#include <cstddef>

// ---------------------------------------------------------------------------
// Problem constants (fixed by the reference: x is (16, 128, 8192) f32)
// ---------------------------------------------------------------------------
#define BB   16
#define CC   128
#define TT   8192
#define K2   (2 * TT)          // interleaved (cos,sin) length in halves
#define LDSPITCH 72            // padded row pitch (halves) for LDS panels

typedef _Float16     v16h __attribute__((ext_vector_type(16)));
typedef _Float16     v8h  __attribute__((ext_vector_type(8)));
typedef float        v8f  __attribute__((ext_vector_type(8)));
typedef int          v4i  __attribute__((ext_vector_type(4)));
typedef int          v8i  __attribute__((ext_vector_type(8)));
typedef unsigned int v4u  __attribute__((ext_vector_type(4)));

#define AS1 __attribute__((address_space(1)))
#define AS3 __attribute__((address_space(3)))

#if defined(__has_builtin)
#  if __has_builtin(__builtin_amdgcn_tensor_load_to_lds) && \
      __has_builtin(__builtin_amdgcn_s_wait_tensorcnt)
#    define HAVE_TDM 1
#  else
#    define HAVE_TDM 0
#  endif
#  if __has_builtin(__builtin_amdgcn_global_load_async_to_lds_b128) && \
      __has_builtin(__builtin_amdgcn_s_wait_asynccnt)
#    define HAVE_ASYNC_LDS 1
#  else
#    define HAVE_ASYNC_LDS 0
#  endif
#else
#  define HAVE_TDM 0
#  define HAVE_ASYNC_LDS 0
#endif

// ---------------------------------------------------------------------------
// Kernel 0: twiddle table tw[m] = (cos, sin)(2*pi*m/8192), m < 4096 (32 KB).
// Computed once per call; read through L2/WGP$ by the FFT kernel.
// ---------------------------------------------------------------------------
__global__ __launch_bounds__(256)
void plv_twiddle_kernel(float2* __restrict__ tw) {
    const int m = blockIdx.x * 256 + threadIdx.x;     // 0..4095
    const float ang = (6.283185307179586f / (float)TT) * (float)m;
    tw[m] = make_float2(cosf(ang), sinf(ang));
}

// ---------------------------------------------------------------------------
// Kernel 1: per-row analytic signal via FFT, output unit phasors as f16.
//   forward DIF (natural in -> bitrev out); the "keep k < T/2" mask becomes
//   "keep even positions" in bitrev order; masked first inverse-DIT stage
//   (s=1, twiddle=1, odd inputs zero) reduces to X[2i+1] = X[2i] (fused);
//   inverse DIT stages s=2..N/2 give natural-order time output. Stage-s
//   twiddle exp(+-i*pi*j/s) = table[j << shl] with shl = log2(4096/s).
//   Scale factors cancel under per-sample unit-magnitude normalization.
// Z layout: row (b*C + c), 2T halves: [cos0, sin0, cos1, sin1, ...]
// ---------------------------------------------------------------------------
__global__ __launch_bounds__(256)
void plv_fft_phase_kernel(const float* __restrict__ x,
                          const float2* __restrict__ tw,
                          _Float16* __restrict__ Z) {
    const int N = TT;
    extern __shared__ float sm[];          // re[8192] | im[8192]  (64 KB)
    float* re = sm;
    float* im = sm + N;

    const int row = blockIdx.x;            // b*C + c
    const int tid = threadIdx.x;
    const float* xr = x + (size_t)row * N;

    for (int i = tid; i < N; i += 256) { re[i] = xr[i]; im[i] = 0.0f; }
    __syncthreads();

    // ---- forward FFT, DIF: butterfly twiddle e^{-i theta} ----
    for (int s = N >> 1, shl = 0; s >= 1; s >>= 1, ++shl) {
        for (int t = tid; t < (N >> 1); t += 256) {
            const int j  = t & (s - 1);
            const int i0 = ((t & ~(s - 1)) << 1) | j;
            const int i1 = i0 + s;
            const float2 w = tw[j << shl];            // (cos th, sin th)
            const float ar = re[i0], ai = im[i0];
            const float br = re[i1], bi = im[i1];
            const float dr = ar - br, di = ai - bi;
            re[i0] = ar + br;  im[i0] = ai + bi;
            re[i1] = dr * w.x + di * w.y;             // (dr+i di) * e^{-i th}
            im[i1] = di * w.x - dr * w.y;
        }
        __syncthreads();
    }

    // ---- fused {mask k>=N/2} + {inverse stage s=1}: X[2i+1] = X[2i] ----
    for (int i = tid; i < (N >> 1); i += 256) {
        re[2 * i + 1] = re[2 * i];
        im[2 * i + 1] = im[2 * i];
    }
    __syncthreads();

    // ---- inverse FFT, DIT stages s=2..N/2: twiddle e^{+i theta} ----
    for (int s = 2, shl = 11; s < N; s <<= 1, --shl) {
        for (int t = tid; t < (N >> 1); t += 256) {
            const int j  = t & (s - 1);
            const int i0 = ((t & ~(s - 1)) << 1) | j;
            const int i1 = i0 + s;
            const float2 w = tw[j << shl];
            const float br = re[i1], bi = im[i1];
            const float tr = br * w.x - bi * w.y;     // (br+i bi) * e^{+i th}
            const float ti = bi * w.x + br * w.y;
            const float ar = re[i0], ai = im[i0];
            re[i0] = ar + tr;  im[i0] = ai + ti;
            re[i1] = ar - tr;  im[i1] = ai - ti;
        }
        __syncthreads();
    }

    // ---- normalize to unit phasor: cos = Re/|z|, sin = Im/|z| ----
    _Float16* zrow = Z + (size_t)row * K2;
    for (int i = tid; i < N; i += 256) {
        const float a = re[i], b = im[i];
        const float m2 = a * a + b * b;
        float c, s;
        if (m2 > 1e-30f) {
            const float r = rsqrtf(m2);
            c = a * r; s = b * r;
        } else {                 // angle(0) == 0 -> cos=1, sin=0
            c = 1.0f; s = 0.0f;
        }
        const _Float16 hc = (_Float16)c;
        const _Float16 hs = (_Float16)s;
        const uint32_t pk = (uint32_t)__builtin_bit_cast(uint16_t, hc)
                          | ((uint32_t)__builtin_bit_cast(uint16_t, hs) << 16);
        *(uint32_t*)(zrow + 2 * i) = pk;   // 4B-aligned coalesced store
    }
}

// ---------------------------------------------------------------------------
// TDM: one descriptor moves a 32-row x 64-half tile (row stride K2 halves in
// memory) into LDS with 4-DWORD padding after every 32 DWORDs -> 72-half
// pitch. D# bit packing per CDNA5 ISA §8.3-8.5 (2-D tensor, groups 2/3 = 0).
// ---------------------------------------------------------------------------
#if HAVE_TDM
__device__ __forceinline__ void tdm_load_panel(const _Float16* gtile,
                                               _Float16* ltile) {
    const uint64_t ga = (uint64_t)(uintptr_t)gtile;
    const uint32_t la = (uint32_t)(uintptr_t)(AS3 _Float16*)ltile;
    v4u g0;
    g0[0] = 1u;                                            // count=1, user mode
    g0[1] = la;                                            // lds_addr (bytes)
    g0[2] = (uint32_t)ga;                                  // global_addr[31:0]
    g0[3] = (uint32_t)((ga >> 32) & 0x01FFFFFFu)           // global_addr[56:32]
          | (2u << 30);                                    // type = 2 ("image")
    v8i g1;
    g1[0] = (int)((1u << 16)        // data_size = 1 -> 2 bytes
                | (1u << 20)        // pad_enable
                | (4u << 22)        // pad_interval: 2^(4+1)=32 DWORDs (=128 B row)
                | (3u << 25));      // pad_amount: 3+1 = 4 DWORDs (=8 halves)
    g1[1] = (int)((uint32_t)K2 << 16);   // tensor_dim0[15:0] @ bits 63:48
    g1[2] = (int)((uint32_t)CC << 16);   // tensor_dim1[15:0] @ bits 95:80
    g1[3] = (int)(64u << 16);            // tile_dim0 = 64  @ bits 127:112
    g1[4] = 32;                          // tile_dim1 = 32  @ bits 143:128
    g1[5] = (int)(uint32_t)K2;           // tensor_dim0_stride[31:0] @ bits 191:160
    g1[6] = 0;
    g1[7] = 0;
    const v4i z4 = {0, 0, 0, 0};
    const v8i z8 = {0, 0, 0, 0, 0, 0, 0, 0};
    // 6-arg toolchain form: (g0, g1, g2, g3, extra group, cpol)
    __builtin_amdgcn_tensor_load_to_lds(g0, g1, z4, z4, z8, 0);
}
#endif

// ---------------------------------------------------------------------------
// Stage one 32x64-half A panel + B panel into LDS (TDM > async-LDS > sync).
// ---------------------------------------------------------------------------
__device__ __forceinline__ void stage_panels(const _Float16* __restrict__ Zb,
                                             int i0, int j0, int kb, int tid,
                                             _Float16* __restrict__ As,
                                             _Float16* __restrict__ Bs) {
#if HAVE_TDM
    if (tid < 32) {                       // wave 0 issues both DMA descriptors
        tdm_load_panel(Zb + (size_t)i0 * K2 + kb, As);
        tdm_load_panel(Zb + (size_t)j0 * K2 + kb, Bs);
    }
#else
#pragma unroll
    for (int q = tid; q < 256; q += 128) {
        const int r  = q >> 3;
        const int ch = q & 7;
        const size_t goff = (size_t)r * K2 + (size_t)kb + (size_t)(ch * 8);
        const _Float16* ga = Zb + (size_t)i0 * K2 + goff;
        const _Float16* gb = Zb + (size_t)j0 * K2 + goff;
        _Float16* la = As + r * LDSPITCH + ch * 8;
        _Float16* lb = Bs + r * LDSPITCH + ch * 8;
#if HAVE_ASYNC_LDS
        __builtin_amdgcn_global_load_async_to_lds_b128(
            (AS1 v4i*)ga, (AS3 v4i*)la, 0, 0);
        __builtin_amdgcn_global_load_async_to_lds_b128(
            (AS1 v4i*)gb, (AS3 v4i*)lb, 0, 0);
#else
        *(v8h*)la = *(const v8h*)ga;
        *(v8h*)lb = *(const v8h*)gb;
#endif
    }
#endif
}

__device__ __forceinline__ void wait_stage_done(int tid) {
#if HAVE_TDM
    if (tid < 32) __builtin_amdgcn_s_wait_tensorcnt(0);   // issuing wave only
#elif HAVE_ASYNC_LDS
    __builtin_amdgcn_s_wait_asynccnt(0);
#endif
    (void)tid;
}

// ---------------------------------------------------------------------------
// Kernel 2: PLV via WMMA, double-buffered LDS fed by TDM/async copies.
//   re[i][j] = sum_K Z_i[K] * Z_j[K]
//   im[i][j] = sum_K W_i[K] * Z_j[K],  W[2t]=Z[2t+1], W[2t+1]=-Z[2t]
//   (W fragment derived in-register: dword swap-halves + sign-flip high half)
// Grid: (C/32, C/32, B). Block: 128 threads = 4 waves in a 2x2 tile grid;
// each wave owns one 16x16 output tile with re+im f32 accumulators.
// ---------------------------------------------------------------------------
__global__ __launch_bounds__(128)
void plv_wmma_kernel(const _Float16* __restrict__ Z, float* __restrict__ out) {
    __shared__ __attribute__((aligned(16))) _Float16 As[2][32][LDSPITCH];
    __shared__ __attribute__((aligned(16))) _Float16 Bs[2][32][LDSPITCH];

    const int b  = blockIdx.z;
    const int i0 = blockIdx.y * 32;
    const int j0 = blockIdx.x * 32;
    const int tid  = threadIdx.x;
    const int lane = tid & 31;
    const int wave = tid >> 5;
    const int wi = wave >> 1;        // row band (0..1) within 32x32 tile
    const int wj = wave & 1;         // col band (0..1)

    const _Float16* Zb = Z + (size_t)b * CC * K2;

    v8f accRe = {};
    v8f accIm = {};

    const int mrow = wi * 16 + (lane & 15);   // A panel row for this lane
    const int nrow = wj * 16 + (lane & 15);   // B panel row (output column)
    const int ksel = lane >> 4;

    // Prologue: stage chunk 0 into buffer 0.
    stage_panels(Zb, i0, j0, 0, tid, &As[0][0][0], &Bs[0][0][0]);
    wait_stage_done(tid);
    __syncthreads();

    int buf = 0;
    for (int kb = 0; kb < K2; kb += 64) {
        // Prefetch next chunk into the other buffer (async; overlap with math).
        if (kb + 64 < K2) {
            stage_panels(Zb, i0, j0, kb + 64, tid,
                         &As[buf ^ 1][0][0], &Bs[buf ^ 1][0][0]);
        }

        const _Float16* Ab = &As[buf][0][0];
        const _Float16* Bb = &Bs[buf][0][0];
#pragma unroll
        for (int ks = 0; ks < 64; ks += 32) {
            // --- A fragment (16x32 f16): lanes 0-15 hold M=lane, K {0..7,16..23};
            //     lanes 16-31 hold M=lane-16, K {8..15,24..31}  (ISA 7.12.2)
            const v8h alo = *(const v8h*)(Ab + mrow * LDSPITCH + ks + ksel * 8);
            const v8h ahi = *(const v8h*)(Ab + mrow * LDSPITCH + ks + 16 + ksel * 8);
            union { v16h h; uint32_t u[8]; } ua, uw;
#pragma unroll
            for (int e = 0; e < 8; ++e) { ua.h[e] = alo[e]; ua.h[e + 8] = ahi[e]; }
            // W fragment: each dword holds an aligned (cos,sin) pair ->
            // (sin, -cos) = swap halves, flip sign bit of high half.
#pragma unroll
            for (int e = 0; e < 8; ++e) {
                const uint32_t z = ua.u[e];
                uw.u[e] = ((z >> 16) | (z << 16)) ^ 0x80000000u;
            }

            // --- B fragment (32x16 f16): lane holds column n=lane&15,
            //     K {0..7,16..23} (low lanes) / K {8..15,24..31} (high lanes)
            const int kk = ks + ksel * 8;
            const v8h blo = *(const v8h*)(Bb + nrow * LDSPITCH + kk);
            const v8h bhi = *(const v8h*)(Bb + nrow * LDSPITCH + kk + 16);
            v16h bfrag;
#pragma unroll
            for (int e = 0; e < 8; ++e) { bfrag[e] = blo[e]; bfrag[e + 8] = bhi[e]; }

            accRe = __builtin_amdgcn_wmma_f32_16x16x32_f16(
                false, ua.h, false, bfrag, (short)0, accRe, false, false);
            accIm = __builtin_amdgcn_wmma_f32_16x16x32_f16(
                false, uw.h, false, bfrag, (short)0, accIm, false, false);
        }

        // Prefetch complete + all waves done reading before buffer reuse.
        wait_stage_done(tid);
        __syncthreads();
        buf ^= 1;
    }

    // Epilogue: D layout — VGPR v: M = v + 8*(lane>>4), N = lane&15.
    const float invT = 1.0f / (float)TT;
    const int col   = j0 + wj * 16 + (lane & 15);
    const int mbase = i0 + wi * 16 + (ksel << 3);
    float* ob = out + (size_t)b * CC * CC;
#pragma unroll
    for (int v = 0; v < 8; ++v) {
        const float r_ = accRe[v];
        const float i_ = accIm[v];
        ob[(size_t)(mbase + v) * CC + col] = invT * sqrtf(r_ * r_ + i_ * i_);
    }
}

// ---------------------------------------------------------------------------
// Launch.  Workspace: Zbuf (64 MB f16 phasors) | twiddle table (32 KB).
// ---------------------------------------------------------------------------
extern "C" void kernel_launch(void* const* d_in, const int* in_sizes, int n_in,
                              void* d_out, int out_size, void* d_ws, size_t ws_size,
                              hipStream_t stream) {
    (void)in_sizes; (void)n_in; (void)out_size; (void)ws_size;
    const float* x  = (const float*)d_in[0];         // (16,128,8192) f32
    float* out      = (float*)d_out;                 // (16,128,128)  f32
    _Float16* Zbuf  = (_Float16*)d_ws;               // (16,128,16384) f16 = 64 MB
    float2* tw      = (float2*)((char*)d_ws + (size_t)BB * CC * K2 * sizeof(_Float16));

    // Kernel 0: 4096-entry twiddle table
    plv_twiddle_kernel<<<dim3(TT / 2 / 256), dim3(256), 0, stream>>>(tw);

    // Kernel 1: 2048 rows, 256 threads, 64 KB dynamic LDS (re+im f32)
    plv_fft_phase_kernel<<<dim3(BB * CC), dim3(256), 2 * TT * sizeof(float), stream>>>(
        x, tw, Zbuf);

    // Kernel 2: (C/32)^2 tiles x B batches = 4*4*16 = 256 workgroups
    plv_wmma_kernel<<<dim3(CC / 32, CC / 32, BB), dim3(128), 0, stream>>>(
        Zbuf, out);
}
